// AttentionBlock_87651692577292
// MI455X (gfx1250) — matrix operand order repaired and checked by reading, not compiled, not run on hardware
//
#include <hip/hip_runtime.h>
#include <hip/hip_bf16.h>
#include <math.h>

typedef __attribute__((ext_vector_type(16))) _Float16 v16h;
typedef __attribute__((ext_vector_type(8)))  float    v8f;

#define D_MODEL 1024
#define D_FF    4096
#define N_HEADS 16
#define D_HEAD  64
#define SEQ_T   2048

// GEMM block tiling
#define BM 128
#define BN 128
#define BK 32

// ---------------------------------------------------------------------------
// f32 -> f16 cast (weights), grid-stride
// ---------------------------------------------------------------------------
__global__ void cast_f32_f16_kernel(const float* __restrict__ in,
                                    _Float16* __restrict__ out, size_t n) {
  size_t i = (size_t)blockIdx.x * blockDim.x + threadIdx.x;
  size_t stride = (size_t)gridDim.x * blockDim.x;
  for (; i < n; i += stride) out[i] = (_Float16)in[i];
}

// ---------------------------------------------------------------------------
// RMSNorm row kernel: out16[row] = x[row] * rsqrt(mean(x^2)+eps) * w, one block/row
// ---------------------------------------------------------------------------
__global__ void rmsnorm_f16_kernel(const float* __restrict__ x,
                                   const float* __restrict__ w,
                                   _Float16* __restrict__ out, int D) {
  int row = blockIdx.x;
  const float* xr = x + (size_t)row * D;
  float ss = 0.f;
  for (int i = threadIdx.x; i < D; i += blockDim.x) { float v = xr[i]; ss += v * v; }
  __shared__ float red[8];
  #pragma unroll
  for (int off = 16; off > 0; off >>= 1) ss += __shfl_down(ss, off, 32);
  int wid = threadIdx.x >> 5, lid = threadIdx.x & 31;
  if (lid == 0) red[wid] = ss;
  __syncthreads();
  if (wid == 0) {
    float v = (lid < (blockDim.x >> 5)) ? red[lid] : 0.f;
    #pragma unroll
    for (int off = 4; off > 0; off >>= 1) v += __shfl_down(v, off, 32);
    if (lid == 0) red[0] = v;
  }
  __syncthreads();
  float scale = rsqrtf(red[0] / (float)D + 1e-6f);
  _Float16* orow = out + (size_t)row * D;
  for (int i = threadIdx.x; i < D; i += blockDim.x)
    orow[i] = (_Float16)(xr[i] * scale * w[i]);
}

// ---------------------------------------------------------------------------
// Stage one 16-half (32B) strip global -> LDS for A and W panels
// ---------------------------------------------------------------------------
__device__ __forceinline__ void stage_tile(const _Float16* __restrict__ Asrc,
                                           const _Float16* __restrict__ Wsrc,
                                           _Float16* Adst, _Float16* Wdst) {
  const uint4* a = (const uint4*)Asrc;
  uint4* ad = (uint4*)Adst;
  ad[0] = a[0]; ad[1] = a[1];
  const uint4* w = (const uint4*)Wsrc;
  uint4* wd = (uint4*)Wdst;
  wd[0] = w[0]; wd[1] = w[1];
}

// ---------------------------------------------------------------------------
// LDS-tiled multi-wave WMMA GEMM: C[M,N] = A16[M,K] @ W16[N,K]^T
//   256 threads = 8 waves; block tile 128x128; wave tile 32x64 (2x4 WMMA tiles)
//   A/W panels (128x32 f16 each) double-buffered in LDS (32 KB total)
// EPI 0: C32[idx] = acc
// EPI 1: C32[idx] = RES[idx] + acc                  (residual add)
// EPI 3: C16[idx] = silu(RES[idx]) * acc            (SwiGLU fuse: RES = gate f32)
// ---------------------------------------------------------------------------
template <int EPI>
__global__ __launch_bounds__(256) void gemm_wmma_tiled_kernel(
    const _Float16* __restrict__ A, const _Float16* __restrict__ W,
    float* __restrict__ C32, _Float16* __restrict__ C16,
    const float* __restrict__ RES, int M, int N, int K) {
  __shared__ _Float16 Alds[2][BM][BK];
  __shared__ _Float16 Wlds[2][BN][BK];

  int nblk = N / BN;
  int bm = blockIdx.x / nblk, bn = blockIdx.x % nblk;
  int gm0 = bm * BM, gn0 = bn * BN;

  int tid = threadIdx.x;
  int lane = tid & 31;
  int half_id = lane >> 4, l16 = lane & 15;
  int wave = tid >> 5;
  int wm = wave >> 1;            // 0..3 -> row offset wm*32
  int wn = wave & 1;             // 0..1 -> col offset wn*64

  // staging assignment: thread -> (row, 16-half strip)
  int srow = tid >> 1;           // 0..127
  int scol = (tid & 1) * 16;     // half-index 0 or 16
  const _Float16* Asrc = A + (size_t)(gm0 + srow) * K + scol;
  const _Float16* Wsrc = W + (size_t)(gn0 + srow) * K + scol;

  v8f acc[2][4];
  #pragma unroll
  for (int mt = 0; mt < 2; mt++)
    #pragma unroll
    for (int nt = 0; nt < 4; nt++) acc[mt][nt] = (v8f){};

  stage_tile(Asrc, Wsrc, &Alds[0][srow][scol], &Wlds[0][srow][scol]);
  __syncthreads();

  for (int k0 = 0; k0 < K; k0 += BK) {
    int buf = (k0 / BK) & 1;
    if (k0 + BK < K)   // prefill next panel while computing this one
      stage_tile(Asrc + k0 + BK, Wsrc + k0 + BK,
                 &Alds[buf ^ 1][srow][scol], &Wlds[buf ^ 1][srow][scol]);
    if (k0 + 2 * BK < K) {       // global_prefetch_b8 two panels ahead
      __builtin_prefetch(Asrc + k0 + 2 * BK, 0, 0);
      __builtin_prefetch(Wsrc + k0 + 2 * BK, 0, 0);
    }

    // A-frag: a[0..15] = K[half*16 .. +15] (two 16B runs)
    v16h afrag[2], bfrag[4];
    #pragma unroll
    for (int mt = 0; mt < 2; mt++) {
      const _Float16* ar = &Alds[buf][wm * 32 + mt * 16 + l16][0];
      ((uint4*)&afrag[mt])[0] = ((const uint4*)(ar + half_id * 16))[0];
      ((uint4*)&afrag[mt])[1] = ((const uint4*)(ar + half_id * 16 + 8))[0];
    }
    // B-frag: b[0..15] = W-row[half*16 .. +15] (32B run)
    #pragma unroll
    for (int nt = 0; nt < 4; nt++) {
      const _Float16* wr = &Wlds[buf][wn * 64 + nt * 16 + l16][half_id * 16];
      ((uint4*)&bfrag[nt])[0] = ((const uint4*)wr)[0];
      ((uint4*)&bfrag[nt])[1] = ((const uint4*)wr)[1];
    }
    #pragma unroll
    for (int mt = 0; mt < 2; mt++)
      #pragma unroll
      for (int nt = 0; nt < 4; nt++)
        acc[mt][nt] = __builtin_amdgcn_wmma_f32_16x16x32_f16(
            false, afrag[mt], false, bfrag[nt], (short)0, acc[mt][nt], false, false);
    __syncthreads();
  }

  // C layout: VGPR r -> M = r + 8*laneHalf, N = lane%16
  #pragma unroll
  for (int mt = 0; mt < 2; mt++)
    #pragma unroll
    for (int nt = 0; nt < 4; nt++)
      #pragma unroll
      for (int r = 0; r < 8; r++) {
        int m = gm0 + wm * 32 + mt * 16 + r + 8 * half_id;
        int n = gn0 + wn * 64 + nt * 16 + l16;
        size_t idx = (size_t)m * N + n;
        float v = acc[mt][nt][r];
        if (EPI == 0) {
          C32[idx] = v;
        } else if (EPI == 1) {
          C32[idx] = RES[idx] + v;
        } else { // EPI == 3
          float g = RES[idx];
          float s = g / (1.f + __expf(-g));
          C16[idx] = (_Float16)(s * v);
        }
      }
}

// ---------------------------------------------------------------------------
// RoPE + split: qkv32 [B*T, 3*1024] -> q16,k16,v16 [B,H,T,Dh] (f16, rope on q,k)
// ---------------------------------------------------------------------------
__global__ void rope_split_kernel(const float* __restrict__ qkv,
                                  _Float16* __restrict__ q16,
                                  _Float16* __restrict__ k16,
                                  _Float16* __restrict__ v16,
                                  int T, size_t total) {
  size_t i = (size_t)blockIdx.x * blockDim.x + threadIdx.x;
  if (i >= total) return;                       // i over B*T*1024
  int col = (int)(i & 1023);
  size_t row = i >> 10;
  int t = (int)(row % T);
  int b = (int)(row / T);
  int h = col >> 6;
  int d = col & 63;
  const float* base = qkv + row * (size_t)(3 * D_MODEL);
  float qv = base[col];
  float kv = base[D_MODEL + col];
  float vv = base[2 * D_MODEL + col];
  int dr = d & 31;
  float sn, cs;
  __sincosf((float)t * __powf(10000.f, -(float)dr / 32.f), &sn, &cs);
  int pair = (d < 32) ? col + 32 : col - 32;
  float qp = base[pair], kp = base[D_MODEL + pair];
  float qrot = (d < 32) ? -qp : qp;     // rotate_half
  float krot = (d < 32) ? -kp : kp;
  size_t o = (((size_t)(b * N_HEADS + h)) * T + t) * D_HEAD + d;
  q16[o] = (_Float16)(qv * cs + qrot * sn);
  k16[o] = (_Float16)(kv * cs + krot * sn);
  v16[o] = (_Float16)vv;
}

// ---------------------------------------------------------------------------
// Flash attention (causal), one wave per (b,h, 16-query tile), 32-key blocks.
// S = Q K^T via 2x wmma (Dh=64); online softmax with shfl_xor row reductions;
// P routed through LDS (C-frag -> A-frag); O += P V via 4x wmma per block.
// Output: attn16 [B, T, H*Dh] f16
// ---------------------------------------------------------------------------
__global__ void flash_attn_wmma_kernel(const _Float16* __restrict__ Q,
                                       const _Float16* __restrict__ K,
                                       const _Float16* __restrict__ V,
                                       _Float16* __restrict__ O, int T) {
  const int qtiles = T >> 4;
  int bh = blockIdx.x / qtiles;
  int qt = blockIdx.x % qtiles;
  int lane = threadIdx.x & 31;
  int half_id = lane >> 4;
  int l16 = lane & 15;

  const _Float16* Qb = Q + (size_t)bh * T * D_HEAD;
  const _Float16* Kb = K + (size_t)bh * T * D_HEAD;
  const _Float16* Vb = V + (size_t)bh * T * D_HEAD;

  // Q A-frags (16 x 64 -> two 16x32 frags), loaded once
  v16h qa[2];
  const _Float16* Qrow = Qb + (size_t)(qt * 16 + l16) * D_HEAD;
  #pragma unroll
  for (int f = 0; f < 2; f++)
    #pragma unroll
    for (int v = 0; v < 8; v++) {
      int kb = f * 32 + half_id * 16 + 2 * v;
      qa[f][2 * v]     = Qrow[kb];
      qa[f][2 * v + 1] = Qrow[kb + 1];
    }

  float rmax[8], rsum[8], corr[8], nmax[8];
  v8f o0 = {}, o1 = {}, o2 = {}, o3 = {};
  #pragma unroll
  for (int r = 0; r < 8; r++) { rmax[r] = -1e30f; rsum[r] = 0.f; }

  __shared__ _Float16 pSh[16 * 32];

  int qhi = qt * 16 + 15;
  int nkb = (qhi >> 5) + 1;                // 32-key blocks covering keys 0..qhi
  for (int kb = 0; kb < nkb; kb++) {
    int key0 = kb * 32;
    // ---- S tile 16x32 as two 16x16 C-frags ----
    v8f s0 = {}, s1 = {};
    #pragma unroll
    for (int sub = 0; sub < 2; sub++) {
      const _Float16* Krow = Kb + (size_t)(key0 + sub * 16 + l16) * D_HEAD;
      #pragma unroll
      for (int f = 0; f < 2; f++) {
        v16h b;
        #pragma unroll
        for (int v = 0; v < 8; v++) {
          int kk = f * 32 + half_id * 16 + 2 * v;
          b[2 * v]     = Krow[kk];
          b[2 * v + 1] = Krow[kk + 1];
        }
        if (sub == 0)
          s0 = __builtin_amdgcn_wmma_f32_16x16x32_f16(false, qa[f], false, b,
                                                      (short)0, s0, false, false);
        else
          s1 = __builtin_amdgcn_wmma_f32_16x16x32_f16(false, qa[f], false, b,
                                                      (short)0, s1, false, false);
      }
    }
    // ---- scale + causal mask + per-row running max ----
    #pragma unroll
    for (int r = 0; r < 8; r++) {
      int qi = qt * 16 + r + 8 * half_id;
      float v0 = s0[r] * 0.125f;           // 1/sqrt(64)
      float v1 = s1[r] * 0.125f;
      if (key0 + l16 > qi)      v0 = -1e30f;
      if (key0 + 16 + l16 > qi) v1 = -1e30f;
      s0[r] = v0; s1[r] = v1;
      float m = fmaxf(rmax[r], fmaxf(v0, v1));
      #pragma unroll
      for (int off = 8; off > 0; off >>= 1) m = fmaxf(m, __shfl_xor(m, off, 32));
      nmax[r] = m;
      corr[r] = __expf(rmax[r] - m);
      rmax[r] = m;
    }
    // ---- P = exp(S - rowmax); row sums ----
    #pragma unroll
    for (int r = 0; r < 8; r++) {
      float p0 = __expf(s0[r] - nmax[r]);
      float p1 = __expf(s1[r] - nmax[r]);
      s0[r] = p0; s1[r] = p1;
      float sum = p0 + p1;
      #pragma unroll
      for (int off = 8; off > 0; off >>= 1) sum += __shfl_xor(sum, off, 32);
      rsum[r] = rsum[r] * corr[r] + sum;
    }
    // ---- C-frag -> A-frag through LDS ----
    #pragma unroll
    for (int r = 0; r < 8; r++) {
      pSh[(r + 8 * half_id) * 32 + l16]      = (_Float16)s0[r];
      pSh[(r + 8 * half_id) * 32 + 16 + l16] = (_Float16)s1[r];
    }
    __syncthreads();
    v16h pa;
    #pragma unroll
    for (int v = 0; v < 8; v++) {
      int kk = half_id * 16 + 2 * v;
      pa[2 * v]     = pSh[l16 * 32 + kk];
      pa[2 * v + 1] = pSh[l16 * 32 + kk + 1];
    }
    __syncthreads();
    // ---- O = O*corr + P @ V  (4 dh-tiles of 16) ----
    const _Float16* Vbase = Vb + (size_t)key0 * D_HEAD + l16;
    #pragma unroll
    for (int dt = 0; dt < 4; dt++) {
      v8f accv = (dt == 0) ? o0 : (dt == 1) ? o1 : (dt == 2) ? o2 : o3;
      #pragma unroll
      for (int r = 0; r < 8; r++) accv[r] *= corr[r];
      v16h vb;
      const _Float16* Vcol = Vbase + dt * 16;
      #pragma unroll
      for (int v = 0; v < 8; v++) {
        int kk = half_id * 16 + 2 * v;
        vb[2 * v]     = Vcol[(size_t)kk * D_HEAD];
        vb[2 * v + 1] = Vcol[(size_t)(kk + 1) * D_HEAD];
      }
      accv = __builtin_amdgcn_wmma_f32_16x16x32_f16(false, pa, false, vb,
                                                    (short)0, accv, false, false);
      if (dt == 0) o0 = accv; else if (dt == 1) o1 = accv;
      else if (dt == 2) o2 = accv; else o3 = accv;
    }
  }
  // ---- normalize + store to [B, T, H*Dh] ----
  int b = bh / N_HEADS, h = bh % N_HEADS;
  #pragma unroll
  for (int dt = 0; dt < 4; dt++) {
    v8f accv = (dt == 0) ? o0 : (dt == 1) ? o1 : (dt == 2) ? o2 : o3;
    #pragma unroll
    for (int r = 0; r < 8; r++) {
      int m = qt * 16 + r + 8 * half_id;
      int dh = dt * 16 + l16;
      O[((size_t)(b * T + m)) * (N_HEADS * D_HEAD) + h * D_HEAD + dh] =
          (_Float16)(accv[r] / rsum[r]);
    }
  }
}

// ---------------------------------------------------------------------------
// Host-side orchestration
// ---------------------------------------------------------------------------
extern "C" void kernel_launch(void* const* d_in, const int* in_sizes, int n_in,
                              void* d_out, int out_size, void* d_ws, size_t ws_size,
                              hipStream_t stream) {
  const float* x     = (const float*)d_in[0];
  const float* wln1  = (const float*)d_in[1];
  const float* wqkv  = (const float*)d_in[2];
  const float* wout  = (const float*)d_in[3];
  const float* wln2  = (const float*)d_in[4];
  const float* wgate = (const float*)d_in[5];
  const float* wup   = (const float*)d_in[6];
  const float* wdown = (const float*)d_in[7];
  float* out = (float*)d_out;

  const int M = in_sizes[0] / D_MODEL;   // B*T = 4096
  const int T = SEQ_T;
  const int B = M / T;

  char* wsp = (char*)d_ws;
  size_t off = 0;
  auto alloc = [&](size_t bytes) -> void* {
    void* p = wsp + off;
    off += (bytes + 255) & ~(size_t)255;
    return p;
  };
  _Float16* h16     = (_Float16*)alloc((size_t)M * D_MODEL * 2);
  _Float16* wqkv16  = (_Float16*)alloc((size_t)3 * D_MODEL * D_MODEL * 2);
  _Float16* wout16  = (_Float16*)alloc((size_t)D_MODEL * D_MODEL * 2);
  _Float16* wgate16 = (_Float16*)alloc((size_t)D_FF * D_MODEL * 2);
  _Float16* wup16   = (_Float16*)alloc((size_t)D_FF * D_MODEL * 2);
  _Float16* wdown16 = (_Float16*)alloc((size_t)D_MODEL * D_FF * 2);
  float*    qkv32   = (float*)alloc((size_t)M * 3 * D_MODEL * 4);
  _Float16* q16     = (_Float16*)alloc((size_t)M * D_MODEL * 2);
  _Float16* k16     = (_Float16*)alloc((size_t)M * D_MODEL * 2);
  _Float16* v16_    = (_Float16*)alloc((size_t)M * D_MODEL * 2);
  _Float16* attn16  = (_Float16*)alloc((size_t)M * D_MODEL * 2);
  float*    x1      = (float*)alloc((size_t)M * D_MODEL * 4);
  float*    g32     = (float*)alloc((size_t)M * D_FF * 4);
  _Float16* mlp16   = (_Float16*)alloc((size_t)M * D_FF * 2);

  // --- weight casts ---
  auto cast = [&](const float* src, _Float16* dst, size_t n) {
    int blocks = (int)((n + 255) / 256);
    if (blocks > 4096) blocks = 4096;
    cast_f32_f16_kernel<<<blocks, 256, 0, stream>>>(src, dst, n);
  };
  cast(wqkv,  wqkv16,  (size_t)3 * D_MODEL * D_MODEL);
  cast(wout,  wout16,  (size_t)D_MODEL * D_MODEL);
  cast(wgate, wgate16, (size_t)D_FF * D_MODEL);
  cast(wup,   wup16,   (size_t)D_FF * D_MODEL);
  cast(wdown, wdown16, (size_t)D_MODEL * D_FF);

  // --- ln1 ---
  rmsnorm_f16_kernel<<<M, 256, 0, stream>>>(x, wln1, h16, D_MODEL);

  // --- qkv = h @ w_qkv^T  [M, 3072] f32 ---
  gemm_wmma_tiled_kernel<0><<<(M / BM) * (3 * D_MODEL / BN), 256, 0, stream>>>(
      h16, wqkv16, qkv32, nullptr, nullptr, M, 3 * D_MODEL, D_MODEL);

  // --- rope + split to [B,H,T,Dh] f16 ---
  {
    size_t total = (size_t)M * D_MODEL;
    rope_split_kernel<<<(int)((total + 255) / 256), 256, 0, stream>>>(
        qkv32, q16, k16, v16_, T, total);
  }

  // --- causal flash attention ---
  flash_attn_wmma_kernel<<<B * N_HEADS * (T / 16), 32, 0, stream>>>(
      q16, k16, v16_, attn16, T);

  // --- x1 = x + attn @ w_out^T ---
  gemm_wmma_tiled_kernel<1><<<(M / BM) * (D_MODEL / BN), 256, 0, stream>>>(
      attn16, wout16, x1, nullptr, x, M, D_MODEL, D_MODEL);

  // --- ln2 (reuse h16) ---
  rmsnorm_f16_kernel<<<M, 256, 0, stream>>>(x1, wln2, h16, D_MODEL);

  // --- gate = h @ w_gate^T  [M, 4096] f32 ---
  gemm_wmma_tiled_kernel<0><<<(M / BM) * (D_FF / BN), 256, 0, stream>>>(
      h16, wgate16, g32, nullptr, nullptr, M, D_FF, D_MODEL);

  // --- mlp16 = silu(gate) * (h @ w_up^T)  [M, 4096] f16 (fused epilogue) ---
  gemm_wmma_tiled_kernel<3><<<(M / BM) * (D_FF / BN), 256, 0, stream>>>(
      h16, wup16, nullptr, mlp16, g32, M, D_FF, D_MODEL);

  // --- out = x1 + mlp16 @ w_down^T ---
  gemm_wmma_tiled_kernel<1><<<(M / BM) * (D_MODEL / BN), 256, 0, stream>>>(
      mlp16, wdown16, out, nullptr, x1, M, D_MODEL, D_FF);
}
